// SABlock_Windows_23845658427969
// MI455X (gfx1250) — matrix operand order repaired and checked by reading, not compiled, not run on hardware
//
#include <hip/hip_runtime.h>
#include <hip/hip_bf16.h>

// ---------------------------------------------------------------------------
// Swin-style window-attention block for MI455X (gfx1250), bf16 WMMA GEMMs.
// ---------------------------------------------------------------------------

typedef __attribute__((ext_vector_type(16))) __bf16 v16bf;
typedef __attribute__((ext_vector_type(8)))  float  v8f;

static constexpr int kB     = 16;
static constexpr int kC     = 384;
static constexpr int kH     = 56;
static constexpr int kW     = 56;
static constexpr int kHeads = 12;
static constexpr int kWS    = 14;
static constexpr int kNTok  = kB * kH * kW;        // 50176
static constexpr int kNWin  = kB * 16;             // 256 windows (4x4 per image)
static constexpr int kNWH   = kNWin * kHeads;      // 3072
static constexpr int kNpad  = 224;                 // 196 padded to 14*16
static constexpr int kMlp   = 4 * kC;              // 1536
static constexpr float kScale = 0.17677669529663687f; // 32^-0.5

// ---------------- WMMA fragment helpers (layouts per CDNA5 ISA 7.12.2) ------

__device__ __forceinline__ v16bf frag_ld_A(const __bf16* p, int klo) {
  // A 16x32 bf16: lanes0-15 K{0-7,16-23}, lanes16-31 K{8-15,24-31}; klo = hi*8
  union { uint4 u[2]; v16bf v; } r;
  r.u[0] = *(const uint4*)(p + klo);
  r.u[1] = *(const uint4*)(p + 16 + klo);
  return r.v;
}
__device__ __forceinline__ v16bf frag_ld_B(const __bf16* p, int hi) {
  // B 32x16 bf16: lane holds column; lanes0-15 K{0-7,16-23}, hi K{8-15,24-31}
  union { uint4 u[2]; v16bf v; } r;
  r.u[0] = *(const uint4*)(p + hi * 8);
  r.u[1] = *(const uint4*)(p + 16 + hi * 8);
  return r.v;
}
__device__ __forceinline__ v8f wmma_bf(v16bf a, v16bf b, v8f c) {
  return __builtin_amdgcn_wmma_f32_16x16x32_bf16(false, a, false, b, (short)0, c,
                                                 false, false);
}

// ---------------- Stage 0: fp32 -> bf16 weight conversion --------------------

__global__ void cvt_bf16_kernel(const float* __restrict__ s, __bf16* __restrict__ d, int n) {
  int i = blockIdx.x * 256 + threadIdx.x;
  if (i < n) d[i] = (__bf16)s[i];
}

// ---------------- Stage 1: depthwise 3x3 conv + bias + residual, NCHW->NHWC --

__global__ __launch_bounds__(256)
void conv_pe_kernel(const float* __restrict__ x, const float* __restrict__ cw,
                    const float* __restrict__ cb, float* __restrict__ xr) {
  __shared__ float t[kW * 193];
  int bh = blockIdx.x;
  int b = bh / kH, h = bh % kH;
  for (int cblk = 0; cblk < kC; cblk += 192) {
    __syncthreads();
    for (int i = threadIdx.x; i < 192 * kW; i += 256) {
      int c = cblk + i / kW, w = i % kW;
      const float* xp = x + ((size_t)b * kC + c) * (kH * kW);
      float acc = xp[h * kW + w] + cb[c];
      const float* wp = cw + c * 9;
#pragma unroll
      for (int dh = -1; dh <= 1; ++dh) {
        int hh2 = h + dh;
        if ((unsigned)hh2 >= (unsigned)kH) continue;
#pragma unroll
        for (int dw = -1; dw <= 1; ++dw) {
          int ww2 = w + dw;
          if ((unsigned)ww2 >= (unsigned)kW) continue;
          acc += xp[hh2 * kW + ww2] * wp[(dh + 1) * 3 + (dw + 1)];
        }
      }
      t[w * 193 + (c - cblk)] = acc;
    }
    __syncthreads();
    for (int i = threadIdx.x; i < kW * 192; i += 256) {
      int w = i / 192, c = i % 192;
      xr[(((size_t)b * kH + h) * kW + w) * kC + cblk + c] = t[w * 193 + c];
    }
  }
}

// ---------------- Stage 2/6: LayerNorm -> bf16 (optionally window-permuted) --

__global__ __launch_bounds__(256)
void ln_kernel(const float* __restrict__ src, __bf16* __restrict__ dst,
               const float* __restrict__ g, const float* __restrict__ beta,
               int windowed) {
  int lane = threadIdx.x & 31, wv = threadIdx.x >> 5;
  int tok = blockIdx.x * 8 + wv;  // destination row index
  int t = tok;
  if (windowed) {
    int win = tok / 196, p = tok - win * 196;
    int bb = win >> 4, wi = win & 15;
    int hh2 = (wi >> 2) * kWS + p / kWS;
    int ww2 = (wi & 3) * kWS + p % kWS;
    t = (bb * kH + hh2) * kW + ww2;
  }
  const float* row = src + (size_t)t * kC;
  float v[12];
  float s = 0.f, s2 = 0.f;
#pragma unroll
  for (int i = 0; i < 12; ++i) {
    float xv = row[lane + i * 32];
    v[i] = xv; s += xv; s2 += xv * xv;
  }
#pragma unroll
  for (int m = 16; m >= 1; m >>= 1) {
    s  += __shfl_xor(s, m);
    s2 += __shfl_xor(s2, m);
  }
  float mu = s * (1.f / kC);
  float var = s2 * (1.f / kC) - mu * mu;
  float rstd = rsqrtf(var + 1e-5f);
  __bf16* drow = dst + (size_t)tok * kC;
#pragma unroll
  for (int i = 0; i < 12; ++i) {
    int c = lane + i * 32;
    drow[c] = (__bf16)((v[i] - mu) * rstd * g[c] + beta[c]);
  }
}

// ---------------- Generic bf16 GEMM: out = A @ W^T, templated epilogue -------
// EPI 0: qkv scatter  1: proj(+bias,+res,permute)  2: fc1(+bias,GELU)  3: fc2(+bias,+res)

template <int EPI>
__global__ __launch_bounds__(256)
void gemm_bf16_kernel(const __bf16* __restrict__ A, const __bf16* __restrict__ W,
                      int M, int N, int K,
                      const float* __restrict__ bias, const float* __restrict__ res,
                      float* __restrict__ outF, __bf16* __restrict__ outB,
                      __bf16* __restrict__ oq, __bf16* __restrict__ ok,
                      __bf16* __restrict__ ov) {
  __shared__ __bf16 As[128][40];
  __shared__ __bf16 Bs[128][40];
  const int tid = threadIdx.x;
  const int n0 = blockIdx.x * 128, m0 = blockIdx.y * 128;
  const int row = tid >> 1, hf = tid & 1;
  const __bf16* ag = A + (size_t)(m0 + row) * K + hf * 16;
  const __bf16* bg = W + (size_t)(n0 + row) * K + hf * 16;

  const int lane = tid & 31, wv = tid >> 5;
  const int wm = wv & 3, wn = wv >> 2;
  const int l16 = lane & 15, hi = lane >> 4, klo = hi * 8;

  v8f acc[2][4] = {};
  const int KT = K >> 5;

  uint4 ra0, ra1, rb0, rb1;
  {
    ra0 = *(const uint4*)(ag);     ra1 = *(const uint4*)(ag + 8);
    rb0 = *(const uint4*)(bg);     rb1 = *(const uint4*)(bg + 8);
  }
  for (int kt = 0; kt < KT; ++kt) {
    *(uint4*)&As[row][hf * 16]     = ra0;
    *(uint4*)&As[row][hf * 16 + 8] = ra1;
    *(uint4*)&Bs[row][hf * 16]     = rb0;
    *(uint4*)&Bs[row][hf * 16 + 8] = rb1;
    __syncthreads();
    if (kt + 1 < KT) {
      const __bf16* ap = ag + (size_t)(kt + 1) * 32;
      const __bf16* bp = bg + (size_t)(kt + 1) * 32;
      ra0 = *(const uint4*)(ap);   ra1 = *(const uint4*)(ap + 8);
      rb0 = *(const uint4*)(bp);   rb1 = *(const uint4*)(bp + 8);
    }
    v16bf a0 = frag_ld_A(&As[wm * 32 + l16][0], klo);
    v16bf a1 = frag_ld_A(&As[wm * 32 + 16 + l16][0], klo);
#pragma unroll
    for (int j = 0; j < 4; ++j) {
      v16bf bj = frag_ld_B(&Bs[wn * 64 + j * 16 + l16][0], hi);
      acc[0][j] = wmma_bf(a0, bj, acc[0][j]);
      acc[1][j] = wmma_bf(a1, bj, acc[1][j]);
    }
    __syncthreads();
  }

#pragma unroll
  for (int i = 0; i < 2; ++i) {
#pragma unroll
    for (int j = 0; j < 4; ++j) {
      const int nb = n0 + wn * 64 + j * 16;
      const int mbase = m0 + wm * 32 + i * 16 + hi * 8;
      if (EPI == 0) {  // qkv scatter: n = s*384 + head*32 + dd
        int sN = nb / 384, rem = nb - sN * 384;
        int head = rem >> 5;
        int dd = (rem & 31) + l16;
#pragma unroll
        for (int r = 0; r < 8; ++r) {
          int tw = mbase + r;
          int win = tw / 196, rr = tw - win * 196;
          size_t whb = (size_t)(win * kHeads + head);
          float v = acc[i][j][r];
          if (sN == 0)      oq[(whb * kNpad + rr) * 32 + dd] = (__bf16)(v * kScale);
          else if (sN == 1) ok[(whb * kNpad + rr) * 32 + dd] = (__bf16)v;
          else              ov[(whb * 32 + dd) * kNpad + rr] = (__bf16)v;
        }
      } else if (EPI == 1) {  // proj: +bias, window-reverse, +residual -> fp32
        int n = nb + l16;
        float bs = bias[n];
#pragma unroll
        for (int r = 0; r < 8; ++r) {
          int tw = mbase + r;
          int win = tw / 196, p = tw - win * 196;
          int bb = win >> 4, wi = win & 15;
          int hh2 = (wi >> 2) * kWS + p / kWS;
          int ww2 = (wi & 3) * kWS + p % kWS;
          size_t t = ((size_t)bb * kH + hh2) * kW + ww2;
          outF[t * kC + n] = res[t * kC + n] + acc[i][j][r] + bs;
        }
      } else if (EPI == 2) {  // fc1: +bias, exact GELU -> bf16
        int n = nb + l16;
        float bs = bias[n];
#pragma unroll
        for (int r = 0; r < 8; ++r) {
          float xv = acc[i][j][r] + bs;
          float gl = 0.5f * xv * (1.0f + erff(xv * 0.70710678118654752f));
          outB[(size_t)(mbase + r) * N + n] = (__bf16)gl;
        }
      } else {  // fc2: +bias, +residual -> fp32
        int n = nb + l16;
        float bs = bias[n];
#pragma unroll
        for (int r = 0; r < 8; ++r) {
          size_t o = (size_t)(mbase + r) * kC + n;
          outF[o] = res[o] + acc[i][j][r] + bs;
        }
      }
    }
  }
}

// ---------------- Stage 4: window attention, one block per (window, head) ----

__global__ __launch_bounds__(128)
void attn_kernel(const __bf16* __restrict__ Q, const __bf16* __restrict__ Kb,
                 const __bf16* __restrict__ Vt, __bf16* __restrict__ O) {
  __shared__ __bf16 ks[kNpad][40];     // k: 224 rows x 32 (padded)
  __shared__ __bf16 vs[32][232];       // v^T: 32 rows x 224 (padded)
  __shared__ __bf16 ps[4][16][232];    // per-wave P strip: 16 x 224 (padded)
  const int wh = blockIdx.x;
  const int tid = threadIdx.x, lane = tid & 31, wv = tid >> 5;
  const __bf16* kg = Kb + (size_t)wh * kNpad * 32;
  const __bf16* vg = Vt + (size_t)wh * 32 * kNpad;
  for (int i = tid; i < kNpad * 4; i += 128) {
    int r = i >> 2, c = (i & 3) * 8;
    *(uint4*)&ks[r][c] = *(const uint4*)(kg + r * 32 + c);
  }
  for (int i = tid; i < 32 * 28; i += 128) {
    int r = i / 28, c = (i % 28) * 8;
    *(uint4*)&vs[r][c] = *(const uint4*)(vg + r * kNpad + c);
  }
  __syncthreads();

  const int l16 = lane & 15, hi = lane >> 4, klo = hi * 8;
  const int win = wh / kHeads, head = wh - win * kHeads;
  const v8f vzero = {};

  for (int s = wv; s < 14; s += 4) {
    const __bf16* qrow = Q + ((size_t)wh * kNpad + s * 16 + l16) * 32;
    v16bf aq = frag_ld_A(qrow, klo);
    v8f S[14];
#pragma unroll
    for (int j = 0; j < 13; ++j) {
      v16bf bk = frag_ld_B(&ks[j * 16 + l16][0], hi);
      S[j] = wmma_bf(aq, bk, vzero);
    }
#pragma unroll
    for (int r = 0; r < 8; ++r) {
      S[13][r] = -1e30f;                                // cols 208..223 all pad
      S[12][r] = (l16 >= 4) ? -1e30f : S[12][r];        // cols 196..207 pad
    }
    float rmax[8], rsum[8];
#pragma unroll
    for (int r = 0; r < 8; ++r) {
      float m = -1e30f;
#pragma unroll
      for (int j = 0; j < 13; ++j) m = fmaxf(m, S[j][r]);
#pragma unroll
      for (int msk = 8; msk >= 1; msk >>= 1) m = fmaxf(m, __shfl_xor(m, msk));
      rmax[r] = m;
    }
#pragma unroll
    for (int r = 0; r < 8; ++r) {
      float sm = 0.f;
#pragma unroll
      for (int j = 0; j < 14; ++j) {
        float p = __expf(S[j][r] - rmax[r]);
        S[j][r] = p;
        sm += p;
      }
#pragma unroll
      for (int msk = 8; msk >= 1; msk >>= 1) sm += __shfl_xor(sm, msk);
      rsum[r] = sm;
    }
    // Round-trip P through LDS to re-fragment as an A-matrix (transpose).
#pragma unroll
    for (int j = 0; j < 14; ++j)
#pragma unroll
      for (int r = 0; r < 8; ++r)
        ps[wv][r + hi * 8][j * 16 + l16] = (__bf16)S[j][r];
    asm volatile("s_wait_dscnt 0" ::: "memory");

    v8f o0 = {}, o1 = {};
#pragma unroll
    for (int kt = 0; kt < 7; ++kt) {
      v16bf ap = frag_ld_A(&ps[wv][l16][kt * 32], klo);
      v16bf b0 = frag_ld_B(&vs[l16][kt * 32], hi);
      v16bf b1 = frag_ld_B(&vs[16 + l16][kt * 32], hi);
      o0 = wmma_bf(ap, b0, o0);
      o1 = wmma_bf(ap, b1, o1);
    }
#pragma unroll
    for (int r = 0; r < 8; ++r) {
      int m = s * 16 + r + hi * 8;
      if (m < 196) {
        size_t tb = ((size_t)win * 196 + m) * kC + head * 32;
        float inv = 1.0f / rsum[r];
        O[tb + l16]      = (__bf16)(o0[r] * inv);
        O[tb + 16 + l16] = (__bf16)(o1[r] * inv);
      }
    }
  }
}

// ---------------- Stage 9: NHWC(fp32) -> NCHW output -------------------------

__global__ __launch_bounds__(256)
void out_transpose_kernel(const float* __restrict__ x2, float* __restrict__ out) {
  __shared__ float t[kW * 193];
  int bh = blockIdx.x;
  int b = bh / kH, h = bh % kH;
  for (int cblk = 0; cblk < kC; cblk += 192) {
    __syncthreads();
    for (int i = threadIdx.x; i < kW * 192; i += 256) {
      int w = i / 192, c = i % 192;
      t[w * 193 + c] = x2[(((size_t)b * kH + h) * kW + w) * kC + cblk + c];
    }
    __syncthreads();
    for (int i = threadIdx.x; i < 192 * kW; i += 256) {
      int c = i / kW, w = i % kW;
      out[(((size_t)b * kC + cblk + c) * kH + h) * kW + w] = t[w * 193 + c];
    }
  }
}

// ---------------- Launcher ---------------------------------------------------

extern "C" void kernel_launch(void* const* d_in, const int* in_sizes, int n_in,
                              void* d_out, int out_size, void* d_ws, size_t ws_size,
                              hipStream_t stream) {
  (void)in_sizes; (void)n_in; (void)out_size; (void)ws_size;
  const float* x      = (const float*)d_in[0];
  const float* conv_w = (const float*)d_in[1];
  const float* conv_b = (const float*)d_in[2];
  const float* ln1_w  = (const float*)d_in[3];
  const float* ln1_b  = (const float*)d_in[4];
  const float* qkv_w  = (const float*)d_in[5];
  const float* proj_w = (const float*)d_in[6];
  const float* proj_b = (const float*)d_in[7];
  const float* ln2_w  = (const float*)d_in[8];
  const float* ln2_b  = (const float*)d_in[9];
  const float* fc1_w  = (const float*)d_in[10];
  const float* fc1_b  = (const float*)d_in[11];
  const float* fc2_w  = (const float*)d_in[12];
  const float* fc2_b  = (const float*)d_in[13];
  float* out = (float*)d_out;

  char* ws = (char*)d_ws;
  size_t off = 0;
  auto alloc = [&](size_t bytes) -> void* {
    void* p = ws + off;
    off = (off + bytes + 255) & ~(size_t)255;
    return p;
  };
  float*  xr   = (float*)alloc((size_t)kNTok * kC * 4);
  float*  x1   = (float*)alloc((size_t)kNTok * kC * 4);
  float*  x2   = (float*)alloc((size_t)kNTok * kC * 4);
  __bf16* xn   = (__bf16*)alloc((size_t)kNTok * kC * 2);
  __bf16* xn2  = (__bf16*)alloc((size_t)kNTok * kC * 2);
  __bf16* aout = (__bf16*)alloc((size_t)kNTok * kC * 2);
  __bf16* hbuf = (__bf16*)alloc((size_t)kNTok * kMlp * 2);
  __bf16* qb   = (__bf16*)alloc((size_t)kNWH * kNpad * 32 * 2);
  __bf16* kb   = (__bf16*)alloc((size_t)kNWH * kNpad * 32 * 2);
  __bf16* vtb  = (__bf16*)alloc((size_t)kNWH * 32 * kNpad * 2);
  __bf16* wq   = (__bf16*)alloc((size_t)3 * kC * kC * 2);
  __bf16* wp   = (__bf16*)alloc((size_t)kC * kC * 2);
  __bf16* w1   = (__bf16*)alloc((size_t)kMlp * kC * 2);
  __bf16* w2   = (__bf16*)alloc((size_t)kC * kMlp * 2);

  // Weight conversions fp32 -> bf16
  {
    int n;
    n = 3 * kC * kC;  cvt_bf16_kernel<<<(n + 255) / 256, 256, 0, stream>>>(qkv_w, wq, n);
    n = kC * kC;      cvt_bf16_kernel<<<(n + 255) / 256, 256, 0, stream>>>(proj_w, wp, n);
    n = kMlp * kC;    cvt_bf16_kernel<<<(n + 255) / 256, 256, 0, stream>>>(fc1_w, w1, n);
    n = kC * kMlp;    cvt_bf16_kernel<<<(n + 255) / 256, 256, 0, stream>>>(fc2_w, w2, n);
  }

  // 1) depthwise conv PE + bias + residual, NCHW -> NHWC fp32
  conv_pe_kernel<<<kB * kH, 256, 0, stream>>>(x, conv_w, conv_b, xr);

  // 2) LN1 -> bf16, window-partition token order
  ln_kernel<<<kNTok / 8, 256, 0, stream>>>(xr, xn, ln1_w, ln1_b, 1);

  // 3) qkv GEMM (M=50176, N=1152, K=384), scatter to q/k/v^T
  gemm_bf16_kernel<0><<<dim3(1152 / 128, kNTok / 128), 256, 0, stream>>>(
      xn, wq, kNTok, 1152, kC, nullptr, nullptr, nullptr, nullptr, qb, kb, vtb);

  // 4) window attention
  attn_kernel<<<kNWH, 128, 0, stream>>>(qb, kb, vtb, aout);

  // 5) proj GEMM + bias + window-reverse + residual -> x1 fp32
  gemm_bf16_kernel<1><<<dim3(kC / 128, kNTok / 128), 256, 0, stream>>>(
      aout, wp, kNTok, kC, kC, proj_b, xr, x1, nullptr, nullptr, nullptr, nullptr);

  // 6) LN2 -> bf16 (token order)
  ln_kernel<<<kNTok / 8, 256, 0, stream>>>(x1, xn2, ln2_w, ln2_b, 0);

  // 7) fc1 GEMM + bias + exact GELU -> bf16 h
  gemm_bf16_kernel<2><<<dim3(kMlp / 128, kNTok / 128), 256, 0, stream>>>(
      xn2, w1, kNTok, kMlp, kC, fc1_b, nullptr, nullptr, hbuf, nullptr, nullptr, nullptr);

  // 8) fc2 GEMM + bias + residual -> x2 fp32
  gemm_bf16_kernel<3><<<dim3(kC / 128, kNTok / 128), 256, 0, stream>>>(
      hbuf, w2, kNTok, kC, kMlp, fc2_b, x1, x2, nullptr, nullptr, nullptr, nullptr);

  // 9) NHWC -> NCHW
  out_transpose_kernel<<<kB * kH, 256, 0, stream>>>(x2, out);
}
